// GreenFunctionNet_61564061221242
// MI455X (gfx1250) — matrix operand order repaired and checked by reading, not compiled, not run on hardware
//
#include <hip/hip_runtime.h>

typedef __attribute__((ext_vector_type(16))) _Float16 v16h;
typedef __attribute__((ext_vector_type(8)))  float    v8f;

#define NROWS 8192
#define PP    512
#define PQ    128           // P quarter handled per wave
#define HA    32
#define HP    32
#define CP    16
#define HG    256

// workspace layout (bytes)
#define OFF_BP    0u        // 512*32 f16  = 32768
#define OFF_WG1T  32768u    // 256*32 f16  = 16384
#define OFF_WG2T  49152u    // 256*256 f16 = 131072
#define OFF_WP1T  180224u   // 32*4 f32    = 512
#define OFF_PST   180736u   // 8192 rows * 4 quarters * 8 f32 = 1048576

static __device__ __forceinline__ float relu1(float x) { return fmaxf(x, 0.0f); }

// lane-half swap (lane <-> lane^16): v_permlanex16_b32 with identity selects
static __device__ __forceinline__ float half_swap(float x) {
#if __has_builtin(__builtin_amdgcn_permlanex16)
  int i = __float_as_int(x);
  i = __builtin_amdgcn_permlanex16(i, i, 0x76543210, 0xfedcba98, false, false);
  return __int_as_float(i);
#else
  return __shfl_xor(x, 16, 32);
#endif
}

// ---------------------------------------------------------------------------
// Prep: f16 tables. bp[p][k] = coeff_x @ Wa1[D:], Wg1T[col][k] (K padded to 32),
// Wg2T[col][k], Wp1T[u][i].
// ---------------------------------------------------------------------------
__global__ __launch_bounds__(256) void prep_kernel(
    const float* __restrict__ cx,  const float* __restrict__ Wa1,
    const float* __restrict__ Wg1, const float* __restrict__ Wg2,
    const float* __restrict__ Wp1,
    _Float16* __restrict__ bp, _Float16* __restrict__ wg1t,
    _Float16* __restrict__ wg2t, float* __restrict__ wp1t) {
  int idx = blockIdx.x * 256 + threadIdx.x;           // 65536 threads total
  { int col = idx >> 8, k = idx & 255;                // Wg2T: [col][k]
    wg2t[idx] = (_Float16)Wg2[k * HG + col]; }
  if (idx < PP * HA) {                                // bp: [p][k]
    int p = idx >> 5, k = idx & 31;
    bp[idx] = (_Float16)(cx[p * 2 + 0] * Wa1[2 * HA + k] +
                         cx[p * 2 + 1] * Wa1[3 * HA + k]);
  }
  if (idx < HG * 32) {                                // Wg1T: [col][k], k>=24 pad 0
    int col = idx >> 5, k = idx & 31;
    wg1t[idx] = (k < 24) ? (_Float16)Wg1[k * HG + col] : (_Float16)0.0f;
  }
  if (idx < 128) {                                    // Wp1T: [u][i]
    int u = idx >> 2, i = idx & 3;
    wp1t[idx] = Wp1[i * HP + u];
  }
}

// ---------------------------------------------------------------------------
// Attention partials: grid 512 blocks x 128 threads (4 waves).
// Block b: p-quarter q = b & 3; wave w handles row tile nTile = (b>>2)*4 + w.
// p-loop unrolled x2: 4 independent WMMAs back-to-back; launch_bounds(128,1)
// relaxes the VGPR cap so each WMMA keeps a distinct D block (no copy-outs,
// hazard windows filled by the other WMMAs / next B build).
// ---------------------------------------------------------------------------
__global__ __launch_bounds__(128, 1) void attn_kernel(
    const float* __restrict__ r,   const float* __restrict__ Wa1,
    const float* __restrict__ Wa2, const float* __restrict__ Wa3,
    const float* __restrict__ cy,  const _Float16* __restrict__ bp,
    float* __restrict__ pst) {
  __shared__ _Float16 lds_bp[PQ * HA];   // 8 KB (this block's quarter)
  __shared__ float4   lds_cy[PQ];        // 2 KB

  int tid = threadIdx.x;
  int q = blockIdx.x & 3;
  int pBase = q * PQ;
  for (int i = tid; i < PQ * HA / 8; i += 128)
    ((float4*)lds_bp)[i] = ((const float4*)(bp + pBase * HA))[i];
  for (int i = tid; i < PQ; i += 128)
    lds_cy[i] = ((const float4*)cy)[pBase + i];
  __syncthreads();

  int wave = tid >> 5, lane = tid & 31, m = lane & 15, h = lane >> 4;
  int nBase = ((blockIdx.x >> 2) * 4 + wave) * 16;
  int n = nBase + m;

  // A-tiles = Wa2^T rows 0..15 (a1) and 16..31 (a2), 16-bit A-layout.
  v16h a1, a2;
#pragma unroll
  for (int e = 0; e < 16; ++e) {
    int k = 16 * h + e;
    a1[e] = (_Float16)Wa2[k * HA + m];
    a2[e] = (_Float16)Wa2[k * HA + 16 + m];
  }
  // a_n[k] for this lane's B-layout slice: k = 16h .. 16h+15
  float q0 = r[n * 4 + 0], q1 = r[n * 4 + 1];
  v16h av;
#pragma unroll
  for (int e = 0; e < 16; ++e) {
    int k = 16 * h + e;
    av[e] = (_Float16)(q0 * Wa1[k] + q1 * Wa1[HA + k]);
  }
  float w3a[8], w3b[8];
#pragma unroll
  for (int v = 0; v < 8; ++v) { w3a[v] = Wa3[v + 8 * h]; w3b[v] = Wa3[16 + v + 8 * h]; }

  const v16h zh = {};
  // two independent online-softmax states (even p / odd p)
  float mrA = -1e30f, srA = 0.f, aA0 = 0.f, aA1 = 0.f, aA2 = 0.f, aA3 = 0.f;
  float mrB = -1e30f, srB = 0.f, aB0 = 0.f, aB1 = 0.f, aB2 = 0.f, aB3 = 0.f;

  for (int p = 0; p < PQ; p += 2) {
    v16h bv0 = *(const v16h*)&lds_bp[(p + 0) * HA + h * 16];
    v16h bv1 = *(const v16h*)&lds_bp[(p + 1) * HA + h * 16];
    v16h bb0 = __builtin_elementwise_max(av + bv0, zh);
    v16h bb1 = __builtin_elementwise_max(av + bv1, zh);
    v8f cz = {};
    v8f d0a = __builtin_amdgcn_wmma_f32_16x16x32_f16(false, a1, false, bb0, (short)0, cz, false, false);
    v8f d0b = __builtin_amdgcn_wmma_f32_16x16x32_f16(false, a2, false, bb0, (short)0, cz, false, false);
    v8f d1a = __builtin_amdgcn_wmma_f32_16x16x32_f16(false, a1, false, bb1, (short)0, cz, false, false);
    v8f d1b = __builtin_amdgcn_wmma_f32_16x16x32_f16(false, a2, false, bb1, (short)0, cz, false, false);

    // Wa3 contraction for both p's: independent accumulator chains.
    float s0 = 0.f, s1 = 0.f, s2 = 0.f, s3 = 0.f;
    float u0 = 0.f, u1 = 0.f, u2 = 0.f, u3 = 0.f;
#pragma unroll
    for (int v = 0; v < 8; v += 4) {
      s0 = fmaf(relu1(d0a[v + 0]), w3a[v + 0], s0);
      s1 = fmaf(relu1(d0a[v + 1]), w3a[v + 1], s1);
      s2 = fmaf(relu1(d0a[v + 2]), w3a[v + 2], s2);
      s3 = fmaf(relu1(d0a[v + 3]), w3a[v + 3], s3);
      s0 = fmaf(relu1(d0b[v + 0]), w3b[v + 0], s0);
      s1 = fmaf(relu1(d0b[v + 1]), w3b[v + 1], s1);
      s2 = fmaf(relu1(d0b[v + 2]), w3b[v + 2], s2);
      s3 = fmaf(relu1(d0b[v + 3]), w3b[v + 3], s3);
      u0 = fmaf(relu1(d1a[v + 0]), w3a[v + 0], u0);
      u1 = fmaf(relu1(d1a[v + 1]), w3a[v + 1], u1);
      u2 = fmaf(relu1(d1a[v + 2]), w3a[v + 2], u2);
      u3 = fmaf(relu1(d1a[v + 3]), w3a[v + 3], u3);
      u0 = fmaf(relu1(d1b[v + 0]), w3b[v + 0], u0);
      u1 = fmaf(relu1(d1b[v + 1]), w3b[v + 1], u1);
      u2 = fmaf(relu1(d1b[v + 2]), w3b[v + 2], u2);
      u3 = fmaf(relu1(d1b[v + 3]), w3b[v + 3], u3);
    }
    float t0 = (s0 + s1) + (s2 + s3);
    float t1 = (u0 + u1) + (u2 + u3);
    t0 += half_swap(t0);                 // logit for p
    t1 += half_swap(t1);                 // logit for p+1

    float4 c0 = lds_cy[p + 0];
    float4 c1 = lds_cy[p + 1];
    // state A <- p (independent of state B <- p+1)
    float mnA = fmaxf(mrA, t0);
    float scA = __expf(mrA - mnA);
    float evA = __expf(t0 - mnA);
    float mnB = fmaxf(mrB, t1);
    float scB = __expf(mrB - mnB);
    float evB = __expf(t1 - mnB);
    srA = srA * scA + evA;
    aA0 = aA0 * scA + evA * c0.x;
    aA1 = aA1 * scA + evA * c0.y;
    aA2 = aA2 * scA + evA * c0.z;
    aA3 = aA3 * scA + evA * c0.w;
    mrA = mnA;
    srB = srB * scB + evB;
    aB0 = aB0 * scB + evB * c1.x;
    aB1 = aB1 * scB + evB * c1.y;
    aB2 = aB2 * scB + evB * c1.z;
    aB3 = aB3 * scB + evB * c1.w;
    mrB = mnB;
  }

  // merge the two states
  float mm = fmaxf(mrA, mrB);
  float wA = __expf(mrA - mm), wB = __expf(mrB - mm);
  float srun = srA * wA + srB * wB;
  float acc0 = aA0 * wA + aB0 * wB;
  float acc1 = aA1 * wA + aB1 * wB;
  float acc2 = aA2 * wA + aB2 * wB;
  float acc3 = aA3 * wA + aB3 * wB;

  if (lane < 16) {
    float* ps = pst + (size_t)(n * 4 + q) * 8;
    float4 lo; lo.x = mm;   lo.y = srun; lo.z = acc0; lo.w = acc1;
    float4 hi; hi.x = acc2; hi.y = acc3; hi.z = 0.f;  hi.w = 0.f;
    *(float4*)ps = lo;
    *(float4*)(ps + 4) = hi;
  }
}

// ---------------------------------------------------------------------------
// Gate MLP: merge 4 softmax partials -> attn_out, c (VALU, tiny),
// g1 = relu(X@Wg1+bg1) via 16 WMMAs -> LDS f16, g2 via 128 WMMAs with two
// interleaved accumulators, Wg3 dot folded per lane, butterfly reduce, exp.
// Grid: 128 blocks x 128 threads (4 waves, 16 rows each).
// ---------------------------------------------------------------------------
__global__ __launch_bounds__(128, 1) void gate_kernel(
    const float* __restrict__ r,   const float* __restrict__ rp,
    const float* __restrict__ pst, const float* __restrict__ wp1t,
    const float* __restrict__ bp1, const float* __restrict__ Wp2,
    const float* __restrict__ bp2, const _Float16* __restrict__ wg1t,
    const float* __restrict__ bg1, const _Float16* __restrict__ wg2t,
    const float* __restrict__ bg2, const float* __restrict__ Wg3,
    const float* __restrict__ bg3, float* __restrict__ out) {
  __shared__ _Float16 g1buf[4 * 16 * HG];   // 32 KB: per-wave [16 rows][256 cols]

  int tid = threadIdx.x;
  int wave = tid >> 5, lane = tid & 31, m = lane & 15, h = lane >> 4;
  int nBase = (blockIdx.x * 4 + wave) * 16;
  int n = nBase + m;

  // ---- merge the 4 online-softmax partials for row n
  const float* ps = pst + (size_t)n * 4 * 8;
  float4 lo0 = *(const float4*)(ps +  0), hi0 = *(const float4*)(ps +  4);
  float4 lo1 = *(const float4*)(ps +  8), hi1 = *(const float4*)(ps + 12);
  float4 lo2 = *(const float4*)(ps + 16), hi2 = *(const float4*)(ps + 20);
  float4 lo3 = *(const float4*)(ps + 24), hi3 = *(const float4*)(ps + 28);
  float mm = fmaxf(fmaxf(lo0.x, lo1.x), fmaxf(lo2.x, lo3.x));
  float w0 = __expf(lo0.x - mm), w1 = __expf(lo1.x - mm);
  float w2 = __expf(lo2.x - mm), w3m = __expf(lo3.x - mm);
  float ssum = lo0.y * w0 + lo1.y * w1 + lo2.y * w2 + lo3.y * w3m;
  float inv = 1.0f / ssum;
  float4 a4;
  a4.x = (lo0.z * w0 + lo1.z * w1 + lo2.z * w2 + lo3.z * w3m) * inv;
  a4.y = (lo0.w * w0 + lo1.w * w1 + lo2.w * w2 + lo3.w * w3m) * inv;
  a4.z = (hi0.x * w0 + hi1.x * w1 + hi2.x * w2 + hi3.x * w3m) * inv;
  a4.w = (hi0.y * w0 + hi1.y * w1 + hi2.y * w2 + hi3.y * w3m) * inv;

  // ---- c = relu(ao@Wp1+bp1)@Wp2+bp2. Half h computes c[jBase..jBase+8).
  int jBase = 8 * (1 - h);
  float cr[8];
#pragma unroll
  for (int jj = 0; jj < 8; ++jj) cr[jj] = bp2[jBase + jj];
  for (int u = 0; u < HP; ++u) {
    float4 w = *(const float4*)&wp1t[u * 4];
    float hv = relu1(bp1[u] + a4.x * w.x + a4.y * w.y + a4.z * w.z + a4.w * w.w);
#pragma unroll
    for (int jj = 0; jj < 8; ++jj) cr[jj] = fmaf(hv, Wp2[u * CP + jBase + jj], cr[jj]);
  }

  // ---- A = X (16 rows x K32, K=24 real + 8 zero pad) in A-layout.
  v16h xa;
#pragma unroll
  for (int e = 0; e < 16; ++e) {
    float xv;
    if (h == 0) {
      if (e < 4)      xv = r[n * 4 + e];
      else if (e < 8) xv = rp[n * 4 + e - 4];
      else            xv = cr[e - 8];
    } else {
      xv = (e < 8) ? cr[e] : 0.f;
    }
    xa[e] = (_Float16)xv;
  }

  // ---- g1 = relu(X @ Wg1 + bg1) -> LDS (f16)
  for (int t = 0; t < 16; ++t) {
    int col = 16 * t + m;
    union { float4 q[2]; v16h v; } ub;
    ub.q[0] = *(const float4*)&wg1t[col * 32 + 8 * h];
    ub.q[1] = *(const float4*)&wg1t[col * 32 + 16 + 8 * h];
    v16h B = ub.v;
    v8f cz = {};
    v8f d = __builtin_amdgcn_wmma_f32_16x16x32_f16(false, xa, false, B, (short)0, cz, false, false);
    float bias = bg1[col];
#pragma unroll
    for (int v = 0; v < 8; ++v) {
      float g = relu1(d[v] + bias);
      g1buf[wave * 4096 + (v + 8 * h) * HG + col] = (_Float16)g;
    }
  }
  __syncthreads();

  // ---- preload g1 A-tiles (8 ktiles of K=32) from LDS in A-layout.
  v16h Ak[8];
#pragma unroll
  for (int kt = 0; kt < 8; ++kt) {
    union { float4 q[2]; v16h v; } u;
    int base = wave * 4096 + m * HG + 32 * kt + 16 * h;
    u.q[0] = *(const float4*)&g1buf[base];
    u.q[1] = *(const float4*)&g1buf[base + 8];
    Ak[kt] = u.v;
  }

  // ---- g2 = relu(g1 @ Wg2 + bg2); fold Wg3 dot into per-lane partials.
  float part[8];
#pragma unroll
  for (int v = 0; v < 8; ++v) part[v] = 0.f;
  float bg3v = bg3[0];

  for (int t2 = 0; t2 < 16; ++t2) {
    int col = 16 * t2 + m;
    v8f dA = {}, dB = {};
#pragma unroll
    for (int kt = 0; kt < 8; kt += 2) {   // two interleaved WMMA accumulators
      v16h B0 = *(const v16h*)&wg2t[col * HG + 32 * (kt + 0) + 16 * h];
      v16h B1 = *(const v16h*)&wg2t[col * HG + 32 * (kt + 1) + 16 * h];
      dA = __builtin_amdgcn_wmma_f32_16x16x32_f16(false, Ak[kt + 0], false, B0, (short)0, dA, false, false);
      dB = __builtin_amdgcn_wmma_f32_16x16x32_f16(false, Ak[kt + 1], false, B1, (short)0, dB, false, false);
    }
    v8f d = dA + dB;
    float bias = bg2[col], w3 = Wg3[col];
#pragma unroll
    for (int v = 0; v < 8; ++v)
      part[v] = fmaf(relu1(d[v] + bias), w3, part[v]);
  }

  // ---- butterfly reduce over 16 lanes per half; part[v] = row (v+8h) sum.
#pragma unroll
  for (int off = 1; off < 16; off <<= 1) {
#pragma unroll
    for (int v = 0; v < 8; ++v) part[v] += __shfl_xor(part[v], off, 32);
  }
  float val = part[0];
#pragma unroll
  for (int v = 1; v < 8; ++v) val = (m == v) ? part[v] : val;
  if (m < 8) out[nBase + m + 8 * h] = __expf(val + bg3v);
}

// ---------------------------------------------------------------------------
extern "C" void kernel_launch(void* const* d_in, const int* in_sizes, int n_in,
                              void* d_out, int out_size, void* d_ws, size_t ws_size,
                              hipStream_t stream) {
  const float* r    = (const float*)d_in[0];
  const float* rprm = (const float*)d_in[1];
  const float* cx   = (const float*)d_in[2];
  const float* cy   = (const float*)d_in[3];
  const float* Wa1  = (const float*)d_in[4];
  const float* Wa2  = (const float*)d_in[5];
  const float* Wa3  = (const float*)d_in[6];
  const float* Wp1  = (const float*)d_in[7];
  const float* bp1  = (const float*)d_in[8];
  const float* Wp2  = (const float*)d_in[9];
  const float* bp2  = (const float*)d_in[10];
  const float* Wg1  = (const float*)d_in[11];
  const float* bg1  = (const float*)d_in[12];
  const float* Wg2  = (const float*)d_in[13];
  const float* bg2  = (const float*)d_in[14];
  const float* Wg3  = (const float*)d_in[15];
  const float* bg3  = (const float*)d_in[16];

  char* ws = (char*)d_ws;
  _Float16* bpT  = (_Float16*)(ws + OFF_BP);
  _Float16* wg1t = (_Float16*)(ws + OFF_WG1T);
  _Float16* wg2t = (_Float16*)(ws + OFF_WG2T);
  float*    wp1t = (float*)(ws + OFF_WP1T);
  float*    pst  = (float*)(ws + OFF_PST);
  float*    out  = (float*)d_out;

  prep_kernel<<<256, 256, 0, stream>>>(cx, Wa1, Wg1, Wg2, Wp1, bpT, wg1t, wg2t, wp1t);
  // 512 row-tiles x 4 p-quarters; 4 waves/block -> 512 blocks
  attn_kernel<<<512, 128, 0, stream>>>(r, Wa1, Wa2, Wa3, cy, bpT, pst);
  // 512 row-tiles, 4 waves/block -> 128 blocks
  gate_kernel<<<128, 128, 0, stream>>>(r, rprm, pst, wp1t, bp1, Wp2, bp2,
                                       wg1t, bg1, wg2t, bg2, Wg3, bg3, out);
}
